// GNNInjectedTransformerLayer_85332410237417
// MI455X (gfx1250) — compile-verified, matrix operand order checked by reading, not run on hardware
//
#include <hip/hip_runtime.h>
#include <math.h>

typedef __bf16 bf16;
typedef bf16  v16bf __attribute__((ext_vector_type(16)));
typedef bf16  v8bf  __attribute__((ext_vector_type(8)));
typedef float v8f   __attribute__((ext_vector_type(8)));

#define B_   8
#define S_   1024
#define D_   512
#define H_   8
#define DFF_ 2048
#define N_   (B_ * S_)
#define DH_  (D_ / H_)

// ---------------------------------------------------------------------------
// CDNA5 async global->LDS copy (16B per lane), tracked by ASYNCcnt.
// Generic pointers to __shared__ carry the LDS offset in their low 32 bits.
// ---------------------------------------------------------------------------
__device__ __forceinline__ void async_b128(void* lds_dst, const bf16* g) {
  unsigned loff = (unsigned)(uintptr_t)lds_dst;
  unsigned long long ga = (unsigned long long)(uintptr_t)g;
  asm volatile("global_load_async_to_lds_b128 %0, %1, off"
               :: "v"(loff), "v"(ga) : "memory");
}
__device__ __forceinline__ void wait_async0() {
  asm volatile("s_wait_asynccnt 0" ::: "memory");
}

// Fragment from LDS (16-bit A/B layout for V_WMMA_*_16X16X32, wave32):
// per lane: v[0..7] = 8 contiguous bf16 at kc, v[8..15] = 8 contiguous at kc+16.
__device__ __forceinline__ v16bf frag_lds(const bf16* q) {
  v8bf lo = *(const v8bf*)(q);
  v8bf hi = *(const v8bf*)(q + 16);
  v16bf v;
#pragma unroll
  for (int i = 0; i < 8; ++i) { v[i] = lo[i]; v[i + 8] = hi[i]; }
  return v;
}

// ---------------------------------------------------------------------------
// bf16 WMMA GEMM with double-buffered async-LDS staging:
//   C[z] = A[z] (MxK) @ W[z]^T + bias     (W is [N,K] row-major)
// Block = 128 threads (4 waves), block tile 64x64, k-step 32.
// Per k-step the block stages A(64x32) and B(64x32) tiles into LDS with
// GLOBAL_LOAD_ASYNC_TO_LDS_B128, waits on ASYNCcnt, and each wave runs a
// 32x32 (2x2 frag) WMMA tile out of LDS. M,N % 64 == 0, K % 32 == 0.
// ---------------------------------------------------------------------------
#define LDSROW 40  // 32 payload + 8 pad bf16 (80B: 16B-aligned, 20-bank stride)

__global__ __launch_bounds__(128)
void gemm_bf16_wmma(const bf16* __restrict__ A, long lda, long saz,
                    const bf16* __restrict__ W, long ldw, long swz,
                    const float* __restrict__ bias,
                    float* __restrict__ Cf, bf16* __restrict__ Cb,
                    long ldc, long scz,
                    int M, int Nn, int K, int do_gelu) {
  __shared__ bf16 Ab[2][64][LDSROW];
  __shared__ bf16 Bb[2][64][LDSROW];

  const int t    = threadIdx.x;
  const int lane = t & 31;
  const int wave = t >> 5;
  const int half = lane >> 4;   // 0: K lo 8, 1: K hi 8
  const int l16  = lane & 15;

  A += (long)blockIdx.z * saz;
  W += (long)blockIdx.z * swz;
  const long cofs  = (long)blockIdx.z * scz;
  const int  mBase = blockIdx.y * 64;
  const int  nBase = blockIdx.x * 64;

  // Cooperative staging: 64 rows x 32 bf16 = 256 x 16B chunks per tile;
  // each thread copies chunks t and t+128 of both tiles.
  auto stage = [&](int kk, int buf) {
#pragma unroll
    for (int c = t; c < 256; c += 128) {
      const int row = c >> 2;
      const int cp  = (c & 3) * 8;
      async_b128(&Ab[buf][row][cp], A + (long)(mBase + row) * lda + kk + cp);
      async_b128(&Bb[buf][row][cp], W + (long)(nBase + row) * ldw + kk + cp);
    }
  };

  v8f acc[2][2];
#pragma unroll
  for (int i = 0; i < 2; ++i)
#pragma unroll
    for (int j = 0; j < 2; ++j)
#pragma unroll
      for (int r = 0; r < 8; ++r) acc[i][j][r] = 0.0f;

  const int ar = (wave >> 1) * 32 + l16;  // A-tile fragment row
  const int br = (wave & 1) * 32 + l16;   // B-tile fragment row (= C column)
  const int kc = half * 8;

  const int nk = K / 32;
  stage(0, 0);
  int p = 0;
  for (int ki = 0; ki < nk; ++ki) {
    wait_async0();        // own async copies into buf p complete
    __syncthreads();      // ... and everyone else's; prior reads of p^1 done
    if (ki + 1 < nk) stage((ki + 1) * 32, p ^ 1);

    v16bf a0 = frag_lds(&Ab[p][ar][kc]);
    v16bf a1 = frag_lds(&Ab[p][ar + 16][kc]);
    v16bf b0 = frag_lds(&Bb[p][br][kc]);
    v16bf b1 = frag_lds(&Bb[p][br + 16][kc]);
    acc[0][0] = __builtin_amdgcn_wmma_f32_16x16x32_bf16(false, a0, false, b0, (short)0, acc[0][0], false, false);
    acc[0][1] = __builtin_amdgcn_wmma_f32_16x16x32_bf16(false, a0, false, b1, (short)0, acc[0][1], false, false);
    acc[1][0] = __builtin_amdgcn_wmma_f32_16x16x32_bf16(false, a1, false, b0, (short)0, acc[1][0], false, false);
    acc[1][1] = __builtin_amdgcn_wmma_f32_16x16x32_bf16(false, a1, false, b1, (short)0, acc[1][1], false, false);
    p ^= 1;
  }

  // Epilogue. C/D frag layout: lanes 0-15 -> M = r, lanes 16-31 -> M = r+8; N = l16.
#pragma unroll
  for (int i = 0; i < 2; ++i) {
#pragma unroll
    for (int j = 0; j < 2; ++j) {
      const int col = nBase + (wave & 1) * 32 + j * 16 + l16;
      const float bv = bias ? bias[col] : 0.0f;
#pragma unroll
      for (int r = 0; r < 8; ++r) {
        const int rowi = mBase + (wave >> 1) * 32 + i * 16 + (half ? r + 8 : r);
        float v = acc[i][j][r] + bv;
        if (do_gelu) v = 0.5f * v * (1.0f + erff(v * 0.70710678118654752f));
        const long idx = cofs + (long)rowi * ldc + col;
        if (Cf) Cf[idx] = v;
        if (Cb) Cb[idx] = (bf16)v;
      }
    }
  }
}

// ---------------------------------------------------------------------------
// Elementwise / reduction helpers
// ---------------------------------------------------------------------------
__global__ void f32_to_bf16(const float* __restrict__ in, bf16* __restrict__ out, long n) {
  long i = (long)blockIdx.x * blockDim.x + threadIdx.x;
  if (i < n) out[i] = (bf16)in[i];
}

__global__ void fill_f32(float* __restrict__ p, float v, long n) {
  long i = (long)blockIdx.x * blockDim.x + threadIdx.x;
  if (i < n) p[i] = v;
}

// Vt[(b*H+h)*DH + d][s] = qkv[(b*S+s)*3D + 2D + h*DH + d]
__global__ void transpose_v(const bf16* __restrict__ qkv, bf16* __restrict__ vt, long total) {
  long i = (long)blockIdx.x * blockDim.x + threadIdx.x;
  if (i >= total) return;
  int  s  = (int)(i % S_);
  long t  = i / S_;
  int  d  = (int)(t % DH_);
  long bh = t / DH_;
  int  h  = (int)(bh % H_);
  int  b  = (int)(bh / H_);
  vt[i] = qkv[((long)(b * S_ + s)) * (3 * D_) + 2 * D_ + h * DH_ + d];
}

// Row softmax over scores[head][q][k] (fp32, one batch) -> probs bf16. scale = 1/sqrt(DH)
__global__ __launch_bounds__(256)
void softmax_rows(const float* __restrict__ s, bf16* __restrict__ p, float scale) {
  const long base = ((long)blockIdx.y * S_ + blockIdx.x) * S_;
  __shared__ float buf[S_];
  __shared__ float red[256];
  float m = -1e30f;
  for (int i = threadIdx.x; i < S_; i += 256) {
    float t = s[base + i] * scale;
    buf[i] = t;
    m = fmaxf(m, t);
  }
  red[threadIdx.x] = m; __syncthreads();
  for (int off = 128; off > 0; off >>= 1) {
    if (threadIdx.x < off) red[threadIdx.x] = fmaxf(red[threadIdx.x], red[threadIdx.x + off]);
    __syncthreads();
  }
  m = red[0]; __syncthreads();
  float sum = 0.0f;
  for (int i = threadIdx.x; i < S_; i += 256) {
    float e = __expf(buf[i] - m);
    buf[i] = e;
    sum += e;
  }
  red[threadIdx.x] = sum; __syncthreads();
  for (int off = 128; off > 0; off >>= 1) {
    if (threadIdx.x < off) red[threadIdx.x] += red[threadIdx.x + off];
    __syncthreads();
  }
  const float inv = 1.0f / red[0];
  for (int i = threadIdx.x; i < S_; i += 256) p[base + i] = (bf16)(buf[i] * inv);
}

// y = LN(a + b + addb; g, beta). Optional fp32 + bf16 outputs. One block per row.
__global__ __launch_bounds__(256)
void ln_fused(const float* __restrict__ a, const float* __restrict__ b,
              const float* __restrict__ addb,
              const float* __restrict__ g, const float* __restrict__ beta,
              float* __restrict__ yf, bf16* __restrict__ yb) {
  const long base = (long)blockIdx.x * D_;
  __shared__ float vals[D_];
  __shared__ float r1[256], r2[256];
  float s = 0.0f, s2 = 0.0f;
  for (int d = threadIdx.x; d < D_; d += 256) {
    float v = a[base + d];
    if (b)    v += b[base + d];
    if (addb) v += addb[d];
    vals[d] = v;
    s += v; s2 += v * v;
  }
  r1[threadIdx.x] = s; r2[threadIdx.x] = s2; __syncthreads();
  for (int off = 128; off > 0; off >>= 1) {
    if (threadIdx.x < off) {
      r1[threadIdx.x] += r1[threadIdx.x + off];
      r2[threadIdx.x] += r2[threadIdx.x + off];
    }
    __syncthreads();
  }
  const float mean = r1[0] * (1.0f / D_);
  const float var  = r2[0] * (1.0f / D_) - mean * mean;
  const float rstd = rsqrtf(var + 1e-5f);
  for (int d = threadIdx.x; d < D_; d += 256) {
    float y = (vals[d] - mean) * rstd * g[d] + beta[d];
    if (yf) yf[base + d] = y;
    if (yb) yb[base + d] = (bf16)y;
  }
}

// ---------------------------------------------------------------------------
// GCN pieces
// ---------------------------------------------------------------------------
__global__ void deg_count(const int* __restrict__ col, float* __restrict__ deg, int E) {
  int e = blockIdx.x * blockDim.x + threadIdx.x;
  if (e < E) atomicAdd(&deg[col[e]], 1.0f);
}

__global__ void to_rsqrt(float* __restrict__ p, int n) {
  int i = blockIdx.x * blockDim.x + threadIdx.x;
  if (i < n) p[i] = rsqrtf(p[i]);
}

// agg init with self-loop contribution: agg[i][d] = xw[i][d] * dinv[i]^2
__global__ void agg_self(const float* __restrict__ xw, const float* __restrict__ dinv,
                         float* __restrict__ agg, long total) {
  long i = (long)blockIdx.x * blockDim.x + threadIdx.x;
  if (i >= total) return;
  const int row = (int)(i / D_);
  const float w = dinv[row] * dinv[row];
  agg[i] = xw[i] * w;
}

// one block per edge: agg[col] += xw[row] * dinv[row]*dinv[col]
__global__ __launch_bounds__(128)
void scatter_edges(const int* __restrict__ er, const int* __restrict__ ec,
                   const float* __restrict__ xw, const float* __restrict__ dinv,
                   float* __restrict__ agg) {
  const int e = blockIdx.x;
  const int r = er[e];
  const int c = ec[e];
  const float w = dinv[r] * dinv[c];
  const float4* __restrict__ src = (const float4*)(xw + (long)r * D_);
  float* __restrict__ dst = agg + (long)c * D_;
  for (int d = threadIdx.x; d < D_ / 4; d += 128) {
    float4 v = src[d];
    atomicAdd(dst + 4 * d + 0, v.x * w);
    atomicAdd(dst + 4 * d + 1, v.y * w);
    atomicAdd(dst + 4 * d + 2, v.z * w);
    atomicAdd(dst + 4 * d + 3, v.w * w);
  }
}

// ---------------------------------------------------------------------------
// Host-side orchestration
// ---------------------------------------------------------------------------
static inline dim3 gemm_grid(int M, int Nn, int Z) { return dim3(Nn / 64, M / 64, Z); }

extern "C" void kernel_launch(void* const* d_in, const int* in_sizes, int n_in,
                              void* d_out, int out_size, void* d_ws, size_t ws_size,
                              hipStream_t stream) {
  const float* src     = (const float*)d_in[0];
  const int*   eidx    = (const int*)d_in[1];
  const float* w_in    = (const float*)d_in[2];
  const float* b_in    = (const float*)d_in[3];
  const float* w_outp  = (const float*)d_in[4];
  const float* b_outp  = (const float*)d_in[5];
  const float* n1_g    = (const float*)d_in[6];
  const float* n1_b    = (const float*)d_in[7];
  const float* w_gcn   = (const float*)d_in[8];
  const float* b_gcn   = (const float*)d_in[9];
  const float* gn_g    = (const float*)d_in[10];
  const float* gn_b    = (const float*)d_in[11];
  const float* w_l1    = (const float*)d_in[12];
  const float* b_l1    = (const float*)d_in[13];
  const float* w_l2    = (const float*)d_in[14];
  const float* b_l2    = (const float*)d_in[15];
  const float* n2_g    = (const float*)d_in[16];
  const float* n2_b    = (const float*)d_in[17];
  float* out = (float*)d_out;

  const int E = in_sizes[1] / 2;
  const int* erow = eidx;       // edge_index[0]
  const int* ecol = eidx + E;   // edge_index[1]

  // ---- workspace carve-up (256B aligned) ----
  size_t off = 0;
  char* base = (char*)d_ws;
  auto alloc = [&](size_t bytes) -> char* {
    char* p = base + off;
    off += (bytes + 255) & ~(size_t)255;
    return p;
  };
  bf16*  xbf    = (bf16*)alloc((size_t)N_ * D_ * 2);
  bf16*  winb   = (bf16*)alloc((size_t)3 * D_ * D_ * 2);
  bf16*  woutb  = (bf16*)alloc((size_t)D_ * D_ * 2);
  bf16*  wgcnb  = (bf16*)alloc((size_t)D_ * D_ * 2);
  bf16*  wl1b   = (bf16*)alloc((size_t)DFF_ * D_ * 2);
  bf16*  wl2b   = (bf16*)alloc((size_t)D_ * DFF_ * 2);
  bf16*  qkv    = (bf16*)alloc((size_t)N_ * 3 * D_ * 2);
  bf16*  vt     = (bf16*)alloc((size_t)N_ * D_ * 2);
  float* scores = (float*)alloc((size_t)H_ * S_ * S_ * 4);   // per-batch reuse
  bf16*  probs  = (bf16*)alloc((size_t)H_ * S_ * S_ * 2);    // per-batch reuse
  bf16*  attnO  = (bf16*)alloc((size_t)N_ * D_ * 2);
  float* attn   = (float*)alloc((size_t)N_ * D_ * 4);
  float* x1     = (float*)alloc((size_t)N_ * D_ * 4);
  bf16*  x1b    = (bf16*)alloc((size_t)N_ * D_ * 2);
  float* xw     = (float*)alloc((size_t)N_ * D_ * 4);
  float* deg    = (float*)alloc((size_t)N_ * 4);
  float* agg    = (float*)alloc((size_t)N_ * D_ * 4);
  float* x2     = (float*)alloc((size_t)N_ * D_ * 4);
  bf16*  x2b    = (bf16*)alloc((size_t)N_ * D_ * 2);
  bf16*  h1     = (bf16*)alloc((size_t)N_ * DFF_ * 2);
  float* ff     = (float*)alloc((size_t)N_ * D_ * 4);
  if (off > ws_size) return;  // workspace too small: bail (harness sizes it)

  const long ND = (long)N_ * D_;
  auto cvt = [&](const float* s_, bf16* d_, long n) {
    f32_to_bf16<<<dim3((unsigned)((n + 255) / 256)), dim3(256), 0, stream>>>(s_, d_, n);
  };

  // ---- 0) bf16 casts of src + all weights ----
  cvt(src,    xbf,   ND);
  cvt(w_in,   winb,  (long)3 * D_ * D_);
  cvt(w_outp, woutb, (long)D_ * D_);
  cvt(w_gcn,  wgcnb, (long)D_ * D_);
  cvt(w_l1,   wl1b,  (long)DFF_ * D_);
  cvt(w_l2,   wl2b,  (long)D_ * DFF_);

  // ---- 1) QKV projection: qkv[N,3D] = src @ in_proj_w^T + b (bf16 out) ----
  gemm_bf16_wmma<<<gemm_grid(N_, 3 * D_, 1), 128, 0, stream>>>(
      xbf, D_, 0, winb, D_, 0, b_in, nullptr, qkv, 3 * D_, 0, N_, 3 * D_, D_, 0);

  // ---- 2) V transpose for the AV GEMM fast path ----
  transpose_v<<<dim3((unsigned)((ND + 255) / 256)), 256, 0, stream>>>(qkv, vt, ND);

  // ---- 3) attention, per batch (scores/probs buffers reused) ----
  const float scale = 1.0f / sqrtf((float)DH_);
  for (int b = 0; b < B_; ++b) {
    const bf16* qb = qkv + (long)b * S_ * 3 * D_;            // head h at col offset h*DH
    const bf16* kb = qb + D_;
    // scores[h][q][k] = Q_h @ K_h^T   (z = head, A/W head stride = DH columns)
    gemm_bf16_wmma<<<gemm_grid(S_, S_, H_), 128, 0, stream>>>(
        qb, 3 * D_, DH_, kb, 3 * D_, DH_, nullptr,
        scores, nullptr, S_, (long)S_ * S_, S_, S_, DH_, 0);
    softmax_rows<<<dim3(S_, H_), 256, 0, stream>>>(scores, probs, scale);
    // O_h[S,DH] = probs_h @ V_h ; Vt gives columns of V contiguously
    gemm_bf16_wmma<<<gemm_grid(S_, DH_, H_), 128, 0, stream>>>(
        probs, S_, (long)S_ * S_, vt + (long)b * H_ * DH_ * S_, S_, (long)DH_ * S_,
        nullptr, nullptr, attnO + (long)b * S_ * D_, D_, DH_, S_, DH_, S_, 0);
  }

  // ---- 4) out projection + residual LN1 ----
  gemm_bf16_wmma<<<gemm_grid(N_, D_, 1), 128, 0, stream>>>(
      attnO, D_, 0, woutb, D_, 0, b_outp, attn, nullptr, D_, 0, N_, D_, D_, 0);
  ln_fused<<<N_, 256, 0, stream>>>(src, attn, nullptr, n1_g, n1_b, x1, x1b);

  // ---- 5) GCN: xw = x1 @ gcn_w^T ; sym-normalized scatter-add ; LN ----
  gemm_bf16_wmma<<<gemm_grid(N_, D_, 1), 128, 0, stream>>>(
      x1b, D_, 0, wgcnb, D_, 0, nullptr, xw, nullptr, D_, 0, N_, D_, D_, 0);
  fill_f32<<<(N_ + 255) / 256, 256, 0, stream>>>(deg, 1.0f, N_);   // self loops
  deg_count<<<(E + 255) / 256, 256, 0, stream>>>(ecol, deg, E);
  to_rsqrt<<<(N_ + 255) / 256, 256, 0, stream>>>(deg, N_);          // deg -> dinv
  agg_self<<<(unsigned)((ND + 255) / 256), 256, 0, stream>>>(xw, deg, agg, ND);
  scatter_edges<<<E, 128, 0, stream>>>(erow, ecol, xw, deg, agg);
  ln_fused<<<N_, 256, 0, stream>>>(x1, agg, b_gcn, gn_g, gn_b, x2, x2b);

  // ---- 6) FFN (exact gelu) + final LN -> d_out ----
  gemm_bf16_wmma<<<gemm_grid(N_, DFF_, 1), 128, 0, stream>>>(
      x2b, D_, 0, wl1b, D_, 0, b_l1, nullptr, h1, DFF_, 0, N_, DFF_, D_, 1);
  gemm_bf16_wmma<<<gemm_grid(N_, D_, 1), 128, 0, stream>>>(
      h1, DFF_, 0, wl2b, DFF_, 0, b_l2, ff, nullptr, D_, 0, N_, D_, DFF_, 0);
  ln_fused<<<N_, 256, 0, stream>>>(x2, ff, nullptr, n2_g, n2_b, out, nullptr);
}
